// GraphSim_21105469293026
// MI455X (gfx1250) — matrix operand order repaired and checked by reading, not compiled, not run on hardware
//
#include <hip/hip_runtime.h>
#include <hip/hip_bf16.h>
#include <math.h>

// ---------------------------------------------------------------------------
// GraphSim on MI455X (gfx1250, wave32).
// Key transform: img = M @ (Xs @ Xt^T) @ M^T  ==  (M@Xs) @ (M@Xt)^T,
// so we never materialize the 8192x8192 similarity matrix.
// All dense GEMMs -> v_wmma_f32_16x16x32_bf16 (one 16x16 tile per wave32).
// Second operand of the image GEMM is stored pre-transposed (K-major) so lanes
// read consecutive N of one K-row per element (coalesced B-fragment loads).
// ---------------------------------------------------------------------------

typedef __bf16 bhalf;
typedef bhalf  bhalf16 __attribute__((ext_vector_type(16)));
typedef float  floatx8 __attribute__((ext_vector_type(8)));

#define NN    8192      // nodes per graph (n1 == n2 == max_n)
#define NE    262144    // edges
#define RSZ   224       // RESHAPE (224 = 14*16, tiles exactly)

// ---------------------------- elementwise helpers ---------------------------

__global__ void to_bf16_kernel(const float* __restrict__ s, bhalf* __restrict__ d, int n) {
    int i = blockIdx.x * blockDim.x + threadIdx.x;
    if (i < n) d[i] = (bhalf)s[i];
}

__global__ void deg_accum_kernel(const long long* __restrict__ tgt,
                                 float* __restrict__ deg, int e_cnt) {
    int e = blockIdx.x * blockDim.x + threadIdx.x;
    if (e < e_cnt) atomicAdd(deg + tgt[e], 1.0f);
}

__global__ void finish_dis_kernel(float* __restrict__ d, int n) {
    int i = blockIdx.x * blockDim.x + threadIdx.x;
    if (i < n) d[i] = rsqrtf(d[i] + 1.0f);     // deg^{-1/2}, deg includes self-loop
}

// out[i][f] = h[i][f] * dis[i]^2   (self-loop term, also initializes accumulator)
__global__ void scale_rows_kernel(const float* __restrict__ h, const float* __restrict__ dis,
                                  float* __restrict__ out, int n, int D) {
    int idx = blockIdx.x * blockDim.x + threadIdx.x;
    if (idx < n * D) {
        int i = idx / D;
        float s = dis[i];
        out[idx] = h[idx] * s * s;
    }
}

// out[tgt] += h[src] * dis[src]*dis[tgt]   (one block per edge, D lanes)
__global__ void edge_prop_kernel(const long long* __restrict__ src,
                                 const long long* __restrict__ tgt,
                                 const float* __restrict__ h,
                                 const float* __restrict__ dis,
                                 float* __restrict__ out, int D) {
    int e = blockIdx.x;
    long long s = src[e], t = tgt[e];
    float ew = dis[s] * dis[t];
    int f = threadIdx.x;
    atomicAdd(out + (size_t)t * D + f, h[(size_t)s * D + f] * ew);
}

__global__ void bias_relu_kernel(float* __restrict__ y, const float* __restrict__ b,
                                 int n, int D) {
    int idx = blockIdx.x * blockDim.x + threadIdx.x;
    if (idx < n * D) y[idx] = fmaxf(y[idx] + b[idx % D], 0.0f);
}

// Bilinear row-resample 8192 -> 224 (align_corners=False); emits bf16 for WMMA.
// trans == 0: dst[r*K + f]  (row-major, A operand)
// trans == 1: dst[f*RSZ + r] (K-major,  B operand of A@B^T as plain K x N)
__global__ void resize_rows_kernel(const float* __restrict__ s, int K,
                                   bhalf* __restrict__ dst, int trans) {
    int r = blockIdx.x;            // 0..223
    int f = threadIdx.x;           // 0..K-1
    float pos = (r + 0.5f) * ((float)NN / (float)RSZ) - 0.5f;
    pos = fminf(fmaxf(pos, 0.0f), (float)(NN - 1));
    int i0 = (int)floorf(pos);
    int i1 = min(i0 + 1, NN - 1);
    float w = pos - (float)i0;
    float v = (1.0f - w) * s[(size_t)i0 * K + f] + w * s[(size_t)i1 * K + f];
    if (trans)
        dst[(size_t)f * RSZ + r] = (bhalf)v;
    else
        dst[(size_t)r * K + f] = (bhalf)v;
}

// ------------------------------- WMMA GEMMs --------------------------------
// C[M,N] = A[M,K] @ B[K,N].  One wave32 per 16x16 C tile, K stepped by 32.
// Fragment layouts follow CDNA5 ISA §7.12.2:
//   A 16x32: lane<16 holds row M=lane, K {0..7,16..23}; lane>=16 K {8..15,24..31}
//   B 32x16: lane holds N-col (lane&15); K {0..7,16..23} lane<16, {8..15,24..31} lane>=16
//   C: VGPR r -> M = (lane>>4)*8 + r, N = lane&15

__global__ __launch_bounds__(32)
void wmma_gemm_kernel(const bhalf* __restrict__ A, const bhalf* __restrict__ B,
                      float* __restrict__ C, int M, int N, int K) {
    int tiles_n = N >> 4;
    int tile = blockIdx.x;
    int tm = tile / tiles_n, tn = tile % tiles_n;
    int lane = threadIdx.x;
    int l  = lane & 15;
    int hi = lane >> 4;

    const bhalf* arow = A + (size_t)(tm * 16 + l) * K;
    floatx8 acc = {};
    for (int kk = 0; kk < K; kk += 32) {
        bhalf16 af, bf;
#pragma unroll
        for (int j = 0; j < 8; ++j) {
            af[j]     = arow[kk + hi * 8 + j];
            af[j + 8] = arow[kk + 16 + hi * 8 + j];
        }
        const bhalf* brow = B + (size_t)(kk + hi * 8) * N + tn * 16 + l;
        __builtin_prefetch(brow + 32 * N, 0, 1);   // global_prefetch_b8 next panel
#pragma unroll
        for (int j = 0; j < 16; ++j) bf[j] = brow[(size_t)(j + (j >= 8 ? 8 : 0)) * N];
        acc = __builtin_amdgcn_wmma_f32_16x16x32_bf16(
            false, af, false, bf, (short)0, acc, false, false);
    }
    int om = tm * 16 + hi * 8;
    int on = tn * 16 + l;
#pragma unroll
    for (int r = 0; r < 8; ++r) C[(size_t)(om + r) * N + on] = acc[r];
}

// C[RSZ,RSZ] = A[RSZ,K] @ BT[K,RSZ]   (similarity image; BT is K-major Mt^T,
// so this is a plain row-major GEMM with fully coalesced fragment loads)
__global__ __launch_bounds__(32)
void wmma_abt_kernel(const bhalf* __restrict__ A, const bhalf* __restrict__ BT,
                     float* __restrict__ C, int K) {
    const int tiles = RSZ >> 4;                 // 14
    int tile = blockIdx.x;
    int tm = tile / tiles, tn = tile % tiles;
    int lane = threadIdx.x;
    int l  = lane & 15;
    int hi = lane >> 4;

    const bhalf* arow = A + (size_t)(tm * 16 + l) * K;
    floatx8 acc = {};
    for (int kk = 0; kk < K; kk += 32) {
        bhalf16 af, bf;
#pragma unroll
        for (int j = 0; j < 8; ++j) {
            af[j]     = arow[kk + hi * 8 + j];
            af[j + 8] = arow[kk + 16 + hi * 8 + j];
        }
        const bhalf* brow = BT + (size_t)(kk + hi * 8) * RSZ + tn * 16 + l;
        __builtin_prefetch(brow + 32 * RSZ, 0, 1);
#pragma unroll
        for (int j = 0; j < 16; ++j) bf[j] = brow[(size_t)(j + (j >= 8 ? 8 : 0)) * RSZ];
        acc = __builtin_amdgcn_wmma_f32_16x16x32_bf16(
            false, af, false, bf, (short)0, acc, false, false);
    }
    int om = tm * 16 + hi * 8;
    int on = tn * 16 + l;
#pragma unroll
    for (int r = 0; r < 8; ++r) C[(size_t)(om + r) * RSZ + on] = acc[r];
}

// ------------------------------- CNN towers --------------------------------

__global__ void conv2d_kernel(const float* __restrict__ in, int Cc, int H, int W,
                              const float* __restrict__ wgt, const float* __restrict__ bias,
                              int OC, int KH, int KW, float* __restrict__ out, int relu) {
    int OH = H - KH + 1, OW = W - KW + 1;
    size_t idx = (size_t)blockIdx.x * blockDim.x + threadIdx.x;
    size_t tot = (size_t)OC * OH * OW;
    if (idx >= tot) return;
    int ox = (int)(idx % OW);
    int t  = (int)(idx / OW);
    int oy = t % OH;
    int oc = t / OH;
    float s = bias[oc];
    for (int ic = 0; ic < Cc; ++ic)
        for (int ky = 0; ky < KH; ++ky) {
            const float* ip = in + ((size_t)ic * H + oy + ky) * W + ox;
            const float* wp = wgt + (((size_t)oc * Cc + ic) * KH + ky) * KW;
#pragma unroll 6
            for (int kx = 0; kx < KW; ++kx) s = fmaf(ip[kx], wp[kx], s);
        }
    out[idx] = relu ? fmaxf(s, 0.0f) : s;
}

__global__ void avgpool_kernel(const float* __restrict__ in, int HW,
                               float* __restrict__ feat, int off) {
    __shared__ float red[256];
    const float* p = in + (size_t)blockIdx.x * HW;
    float s = 0.0f;
    for (int i = threadIdx.x; i < HW; i += blockDim.x) s += p[i];
    red[threadIdx.x] = s;
    __syncthreads();
    for (int st = blockDim.x >> 1; st > 0; st >>= 1) {
        if ((int)threadIdx.x < st) red[threadIdx.x] += red[threadIdx.x + st];
        __syncthreads();
    }
    if (threadIdx.x == 0) feat[off + blockIdx.x] = red[0] / (float)HW;
}

// ------------------------------- MLP head ----------------------------------

__global__ __launch_bounds__(64)
void mlp_head_kernel(const float* __restrict__ feat,
                     const float* __restrict__ l1w, const float* __restrict__ l1b,
                     const float* __restrict__ l2w, const float* __restrict__ l2b,
                     const float* __restrict__ l3w, const float* __restrict__ l3b,
                     float* __restrict__ out) {
    __shared__ float h1[64], h2[64];
    int i = threadIdx.x;
    float s = l1b[i];
    for (int k = 0; k < 96; ++k) s = fmaf(feat[k], l1w[k * 64 + i], s);
    h1[i] = fmaxf(s, 0.0f);
    __syncthreads();
    s = l2b[i];
    for (int k = 0; k < 64; ++k) s = fmaf(h1[k], l2w[k * 64 + i], s);
    h2[i] = fmaxf(s, 0.0f);
    __syncthreads();
    if (i == 0) {
        float t = l3b[0];
        for (int k = 0; k < 64; ++k) t = fmaf(h2[k], l3w[k], t);
        out[0] = 1.0f / (1.0f + expf(-t));
    }
}

// ------------------------------- host driver -------------------------------

extern "C" void kernel_launch(void* const* d_in, const int* in_sizes, int n_in,
                              void* d_out, int out_size, void* d_ws, size_t ws_size,
                              hipStream_t stream) {
    (void)in_sizes; (void)n_in; (void)out_size; (void)ws_size;

    // setup_inputs() dict order
    const float*     x1  = (const float*)d_in[0];
    const long long* ei1 = (const long long*)d_in[1];   // [2, E] int64
    const float*     x2  = (const float*)d_in[2];
    const long long* ei2 = (const long long*)d_in[3];
    // params (insertion order): W1,b1,W2,b2,W3,b3, cnn0{w1,b1..w4,b4}, cnn1, cnn2, l1w..l3b
    const float* W[3]  = {(const float*)d_in[4], (const float*)d_in[6], (const float*)d_in[8]};
    const float* Bv[3] = {(const float*)d_in[5], (const float*)d_in[7], (const float*)d_in[9]};
    const float* l1w = (const float*)d_in[34];
    const float* l1b = (const float*)d_in[35];
    const float* l2w = (const float*)d_in[36];
    const float* l2b = (const float*)d_in[37];
    const float* l3w = (const float*)d_in[38];
    const float* l3b = (const float*)d_in[39];

    const int Dly[3] = {128, 64, 32};     // layer output dims
    const int Kly[3] = {128, 128, 64};    // layer input dims

    // ---- workspace carve ----
    char* wsp = (char*)d_ws;
    auto carve = [&](size_t bytes) -> void* {
        void* p = (void*)wsp;
        wsp += (bytes + 255) & ~(size_t)255;
        return p;
    };
    float* dis[2]  = {(float*)carve(NN * 4), (float*)carve(NN * 4)};
    bhalf* xbf     = (bhalf*)carve((size_t)NN * 128 * 2);
    bhalf* wbf     = (bhalf*)carve(128 * 128 * 2);
    float* hw      = (float*)carve((size_t)NN * 128 * 4);
    float* emb[2][3];
    for (int g = 0; g < 2; ++g)
        for (int j = 0; j < 3; ++j) emb[g][j] = (float*)carve((size_t)NN * Dly[j] * 4);
    // graph 0: row-major (A operand); graph 1: K-major transposed (B operand)
    bhalf* ms[2][3];
    for (int g = 0; g < 2; ++g)
        for (int j = 0; j < 3; ++j) ms[g][j] = (bhalf*)carve((size_t)RSZ * Dly[j] * 2);
    float* img   = (float*)carve((size_t)RSZ * RSZ * 4);
    float* cbuf0 = (float*)carve((size_t)32 * 219 * 219 * 4);
    float* cbuf1 = (float*)carve((size_t)32 * 219 * 219 * 4);
    float* feat  = (float*)carve(96 * 4);

    // ---- GCN towers for both graphs ----
    for (int g = 0; g < 2; ++g) {
        const float*     x    = (g == 0) ? x1  : x2;
        const long long* ei   = (g == 0) ? ei1 : ei2;
        const long long* esrc = ei;
        const long long* etgt = ei + NE;

        hipMemsetAsync(dis[g], 0, NN * sizeof(float), stream);
        deg_accum_kernel<<<(NE + 255) / 256, 256, 0, stream>>>(etgt, dis[g], NE);
        finish_dis_kernel<<<(NN + 255) / 256, 256, 0, stream>>>(dis[g], NN);

        const float* cur = x;
        for (int j = 0; j < 3; ++j) {
            int Kd = Kly[j], Dd = Dly[j];
            int nin = NN * Kd, nw = Kd * Dd, nout = NN * Dd;
            to_bf16_kernel<<<(nin + 255) / 256, 256, 0, stream>>>(cur, xbf, nin);
            to_bf16_kernel<<<(nw + 255) / 256, 256, 0, stream>>>(W[j], wbf, nw);
            wmma_gemm_kernel<<<(NN / 16) * (Dd / 16), 32, 0, stream>>>(
                xbf, wbf, hw, NN, Dd, Kd);
            scale_rows_kernel<<<(nout + 255) / 256, 256, 0, stream>>>(
                hw, dis[g], emb[g][j], NN, Dd);
            edge_prop_kernel<<<NE, Dd, 0, stream>>>(esrc, etgt, hw, dis[g], emb[g][j], Dd);
            bias_relu_kernel<<<(nout + 255) / 256, 256, 0, stream>>>(
                emb[g][j], Bv[j], NN, Dd);
            cur = emb[g][j];
        }
        // resampled embeddings (M @ emb); graph 1 emitted transposed for the GEMM
        for (int j = 0; j < 3; ++j)
            resize_rows_kernel<<<RSZ, Dly[j], 0, stream>>>(
                emb[g][j], Dly[j], ms[g][j], g);
    }

    // ---- similarity images + CNN towers ----
    for (int j = 0; j < 3; ++j) {
        wmma_abt_kernel<<<(RSZ / 16) * (RSZ / 16), 32, 0, stream>>>(
            ms[0][j], ms[1][j], img, Dly[j]);

        const int pb = 10 + j * 8;   // cnnj params: w1,b1,w2,b2,w3,b3,w4,b4
        const float* cw[4] = {(const float*)d_in[pb],     (const float*)d_in[pb + 2],
                              (const float*)d_in[pb + 4], (const float*)d_in[pb + 6]};
        const float* cb[4] = {(const float*)d_in[pb + 1], (const float*)d_in[pb + 3],
                              (const float*)d_in[pb + 5], (const float*)d_in[pb + 7]};

        // conv1: 1x224x224 -> 8x219x219
        {   size_t tot = (size_t)8 * 219 * 219;
            conv2d_kernel<<<(tot + 255) / 256, 256, 0, stream>>>(
                img, 1, 224, 224, cw[0], cb[0], 8, 6, 6, cbuf0, 1); }
        // conv2: 8x219x219 -> 16x214x214
        {   size_t tot = (size_t)16 * 214 * 214;
            conv2d_kernel<<<(tot + 255) / 256, 256, 0, stream>>>(
                cbuf0, 8, 219, 219, cw[1], cb[1], 16, 6, 6, cbuf1, 1); }
        // conv3: 16x214x214 -> 32x210x210
        {   size_t tot = (size_t)32 * 210 * 210;
            conv2d_kernel<<<(tot + 255) / 256, 256, 0, stream>>>(
                cbuf1, 16, 214, 214, cw[2], cb[2], 32, 5, 5, cbuf0, 1); }
        // conv4: 32x210x210 -> 32x206x206 (no relu)
        {   size_t tot = (size_t)32 * 206 * 206;
            conv2d_kernel<<<(tot + 255) / 256, 256, 0, stream>>>(
                cbuf0, 32, 210, 210, cw[3], cb[3], 32, 5, 5, cbuf1, 0); }
        avgpool_kernel<<<32, 256, 0, stream>>>(cbuf1, 206 * 206, feat, j * 32);
    }

    mlp_head_kernel<<<1, 64, 0, stream>>>(feat, l1w, l1b, l2w, l2b, l3w, l3b,
                                          (float*)d_out);
}
